// AetheriumTransformerBlock_81844896792617
// MI455X (gfx1250) — compile-verified, matrix operand order checked by reading, not run on hardware
//
#include <hip/hip_runtime.h>
#include <hip/hip_bf16.h>

typedef __bf16 bf16_t;
typedef __bf16 v16bf __attribute__((ext_vector_type(16)));
typedef __bf16 v8bf  __attribute__((ext_vector_type(8)));
typedef float  v8f   __attribute__((ext_vector_type(8)));

constexpr int S     = 1024;
constexpr int Hdim  = 1024;
constexpr int HDh   = 256;   // head dim == LAT
constexpr int NHEAD = 4;
constexpr int NEXP  = 8;
constexpr float EPSf = 1e-5f;

// ---------- helpers ----------
__device__ __forceinline__ v16bf combine16(v8bf lo, v8bf hi) {
  v16bf r;
#pragma unroll
  for (int i = 0; i < 8; ++i) { r[i] = lo[i]; r[i + 8] = hi[i]; }
  return r;
}
__device__ __forceinline__ v8bf loadW8(const float* p) {
  v8f t = *(const v8f*)p;
  v8bf r;
#pragma unroll
  for (int i = 0; i < 8; ++i) r[i] = (bf16_t)t[i];
  return r;
}
__device__ __forceinline__ v8bf loadW8(const bf16_t* p) {
  return *(const v8bf*)p;
}

// ---------- fp32 -> bf16 streaming conversion (weight cache) ----------
__global__ __launch_bounds__(256) void cvt_bf16_kernel(
    const float* __restrict__ src, bf16_t* __restrict__ dst, long n)
{
  const long i = ((long)blockIdx.x * 256 + threadIdx.x) * 8;
  if (i + 8 <= n) {
    v8f t = *(const v8f*)(src + i);
    v8bf r;
#pragma unroll
    for (int k = 0; k < 8; ++k) r[k] = (bf16_t)t[k];
    *(v8bf*)(dst + i) = r;
  }
}

// ---------- generic WMMA GEMM: C = A[M,K](bf16) @ W[N,K]^T ----------
// block = 256 thr = 8 waves; wave tile 32(M) x 64(N); block tile 64 x 256.
enum { EPI_F32 = 0, EPI_BF16 = 1, EPI_RES = 2, EPI_DOWN = 3, EPI_TANH = 4 };

template <int EPI, typename WT>
__global__ __launch_bounds__(256) void gemm_kernel(
    const bf16_t* __restrict__ A, int lda, long aBatch,
    const WT* __restrict__ W, int ldw, long wBatch,
    void* __restrict__ Cv, int ldc, long cBatch,
    int K, float scale,
    const float* __restrict__ R, int ldr,
    const float* __restrict__ tw, int twStride)
{
  A += (long)blockIdx.z * aBatch;
  W += (long)blockIdx.z * wBatch;
  const int lane = threadIdx.x & 31;
  const int wave = threadIdx.x >> 5;
  const int l16  = lane & 15;
  const int half = lane >> 4;
  const int tileM = blockIdx.y * 64 + (wave & 1) * 32;
  const int tileN = blockIdx.x * 256 + (wave >> 1) * 64;

  v8f acc[2][4] = {};
  const bf16_t* aRow0 = A + (long)(tileM + l16) * lda;
  const bf16_t* aRow1 = A + (long)(tileM + 16 + l16) * lda;
  const WT* wRow[4];
#pragma unroll
  for (int j = 0; j < 4; ++j) wRow[j] = W + (long)(tileN + j * 16 + l16) * ldw;

  for (int k0 = 0; k0 < K; k0 += 32) {
    const int ka = k0 + half * 8;
    v16bf av0 = combine16(*(const v8bf*)(aRow0 + ka), *(const v8bf*)(aRow0 + ka + 16));
    v16bf av1 = combine16(*(const v8bf*)(aRow1 + ka), *(const v8bf*)(aRow1 + ka + 16));
#pragma unroll
    for (int j = 0; j < 4; ++j) {
      v16bf bv = combine16(loadW8(wRow[j] + ka), loadW8(wRow[j] + ka + 16));
      acc[0][j] = __builtin_amdgcn_wmma_f32_16x16x32_bf16(
          false, av0, false, bv, (short)0, acc[0][j], false, false);
      acc[1][j] = __builtin_amdgcn_wmma_f32_16x16x32_bf16(
          false, av1, false, bv, (short)0, acc[1][j], false, false);
    }
  }

  float*  Cf = (float*)Cv  + (long)blockIdx.z * cBatch;
  bf16_t* Cb = (bf16_t*)Cv + (long)blockIdx.z * cBatch;
#pragma unroll
  for (int i = 0; i < 2; ++i) {
#pragma unroll
    for (int j = 0; j < 4; ++j) {
#pragma unroll
      for (int r = 0; r < 8; ++r) {
        const int m = tileM + i * 16 + half * 8 + r;
        const int n = tileN + j * 16 + l16;
        const float v = acc[i][j][r] * scale;
        if (EPI == EPI_F32) {
          Cf[(long)m * ldc + n] = v;
        } else if (EPI == EPI_BF16) {
          Cb[(long)m * ldc + n] = (bf16_t)v;
        } else if (EPI == EPI_RES) {
          Cf[(long)m * ldc + n] = v + R[(long)m * ldr + n];
        } else if (EPI == EPI_DOWN) {
          Cf[(long)m * ldc + n] += tw[m * twStride] * v;
        } else if (EPI == EPI_TANH) {
          Cf[(long)m * ldc + n] += tw[m * twStride] * 0.2f * tanhf(v);
        }
      }
    }
  }
}

// ---------- fused gate/up kernel: out = silu(A@Wg^T) * (A@Wu^T) * qscale ----------
template <typename WT>
__global__ __launch_bounds__(256) void gateup_kernel(
    const bf16_t* __restrict__ A, const WT* __restrict__ Wg,
    const WT* __restrict__ Wu, bf16_t* __restrict__ Out,
    int N, int K, float qscale)
{
  const int lane = threadIdx.x & 31;
  const int wave = threadIdx.x >> 5;
  const int l16  = lane & 15;
  const int half = lane >> 4;
  const int tileM = blockIdx.y * 64 + (wave & 1) * 32;
  const int tileN = blockIdx.x * 256 + (wave >> 1) * 64;

  v8f accG[2][4] = {};
  v8f accU[2][4] = {};
  const bf16_t* aRow0 = A + (long)(tileM + l16) * K;
  const bf16_t* aRow1 = A + (long)(tileM + 16 + l16) * K;
  const WT* gRow[4];
  const WT* uRow[4];
#pragma unroll
  for (int j = 0; j < 4; ++j) {
    gRow[j] = Wg + (long)(tileN + j * 16 + l16) * K;
    uRow[j] = Wu + (long)(tileN + j * 16 + l16) * K;
  }

  for (int k0 = 0; k0 < K; k0 += 32) {
    const int ka = k0 + half * 8;
    v16bf av0 = combine16(*(const v8bf*)(aRow0 + ka), *(const v8bf*)(aRow0 + ka + 16));
    v16bf av1 = combine16(*(const v8bf*)(aRow1 + ka), *(const v8bf*)(aRow1 + ka + 16));
#pragma unroll
    for (int j = 0; j < 4; ++j) {
      v16bf bg = combine16(loadW8(gRow[j] + ka), loadW8(gRow[j] + ka + 16));
      accG[0][j] = __builtin_amdgcn_wmma_f32_16x16x32_bf16(
          false, av0, false, bg, (short)0, accG[0][j], false, false);
      accG[1][j] = __builtin_amdgcn_wmma_f32_16x16x32_bf16(
          false, av1, false, bg, (short)0, accG[1][j], false, false);
      v16bf bu = combine16(loadW8(uRow[j] + ka), loadW8(uRow[j] + ka + 16));
      accU[0][j] = __builtin_amdgcn_wmma_f32_16x16x32_bf16(
          false, av0, false, bu, (short)0, accU[0][j], false, false);
      accU[1][j] = __builtin_amdgcn_wmma_f32_16x16x32_bf16(
          false, av1, false, bu, (short)0, accU[1][j], false, false);
    }
  }

#pragma unroll
  for (int i = 0; i < 2; ++i) {
#pragma unroll
    for (int j = 0; j < 4; ++j) {
#pragma unroll
      for (int r = 0; r < 8; ++r) {
        const int m = tileM + i * 16 + half * 8 + r;
        const int n = tileN + j * 16 + l16;
        float g = accG[i][j][r];
        float u = accU[i][j][r];
        float sig = 1.f / (1.f + expf(-g));
        Out[(long)m * N + n] = (bf16_t)(g * sig * u * qscale);
      }
    }
  }
}

// ---------- layernorm (writes bf16 + optional fp32) ----------
__global__ __launch_bounds__(256) void ln_kernel(
    const float* __restrict__ x, const float* __restrict__ w,
    const float* __restrict__ b, bf16_t* __restrict__ outb,
    float* __restrict__ outf)
{
  const int row = blockIdx.x, tid = threadIdx.x;
  __shared__ float r1[256], r2[256];
  const float* xr = x + (long)row * Hdim;
  float v[4];
  float s = 0.f, sq = 0.f;
#pragma unroll
  for (int i = 0; i < 4; ++i) {
    v[i] = xr[tid + 256 * i];
    s += v[i]; sq += v[i] * v[i];
  }
  r1[tid] = s; r2[tid] = sq; __syncthreads();
  for (int o = 128; o > 0; o >>= 1) {
    if (tid < o) { r1[tid] += r1[tid + o]; r2[tid] += r2[tid + o]; }
    __syncthreads();
  }
  const float mean = r1[0] * (1.f / Hdim);
  const float var  = r2[0] * (1.f / Hdim) - mean * mean;
  const float inv  = rsqrtf(var + EPSf);
#pragma unroll
  for (int i = 0; i < 4; ++i) {
    const int c = tid + 256 * i;
    float y = (v[i] - mean) * inv * w[c] + b[c];
    outb[(long)row * Hdim + c] = (bf16_t)y;
    if (outf) outf[(long)row * Hdim + c] = y;
  }
}

// ---------- RoPE(q,k) -> bf16, and V -> V^T bf16 ----------
__global__ __launch_bounds__(256) void ropev_kernel(
    const float* __restrict__ qf, const float* __restrict__ kf,
    const float* __restrict__ vf, bf16_t* __restrict__ qb,
    bf16_t* __restrict__ kb, bf16_t* __restrict__ vTb)
{
  const int s = blockIdx.x, tid = threadIdx.x;
  for (int i = tid; i < NHEAD * 128; i += 256) {
    const int head = i >> 7, fi = i & 127;
    const float freq = powf(10000.f, -(2.f * fi) / (float)HDh);
    const float ang = (float)s * freq;
    const float c = cosf(ang), sn = sinf(ang);
    const long idx = (long)s * Hdim + head * HDh + 2 * fi;
    const float a = qf[idx], b = qf[idx + 1];
    qb[idx]     = (bf16_t)(a * c - b * sn);
    qb[idx + 1] = (bf16_t)(b * c + a * sn);
  }
  for (int i = tid; i < 128; i += 256) {
    const float freq = powf(10000.f, -(2.f * i) / (float)HDh);
    const float ang = (float)s * freq;
    const float c = cosf(ang), sn = sinf(ang);
    const long idx = (long)s * HDh + 2 * i;
    const float a = kf[idx], b = kf[idx + 1];
    kb[idx]     = (bf16_t)(a * c - b * sn);
    kb[idx + 1] = (bf16_t)(b * c + a * sn);
  }
  for (int d = tid; d < HDh; d += 256)
    vTb[(long)d * S + s] = (bf16_t)vf[(long)s * HDh + d];
}

// ---------- softmax over attention rows (fp32 in, bf16 out) ----------
__global__ __launch_bounds__(256) void softmax_kernel(
    const float* __restrict__ sc, bf16_t* __restrict__ at)
{
  const int rid = blockIdx.x, tid = threadIdx.x;
  const float* src = sc + (long)rid * S;
  bf16_t* dst = at + (long)rid * S;
  __shared__ float red[256];
  float v[4];
  float lm = -3.4e38f;
#pragma unroll
  for (int i = 0; i < 4; ++i) { v[i] = src[tid + 256 * i]; lm = fmaxf(lm, v[i]); }
  red[tid] = lm; __syncthreads();
  for (int o = 128; o > 0; o >>= 1) {
    if (tid < o) red[tid] = fmaxf(red[tid], red[tid + o]);
    __syncthreads();
  }
  const float m = red[0]; __syncthreads();
  float ls = 0.f;
#pragma unroll
  for (int i = 0; i < 4; ++i) { v[i] = expf(v[i] - m); ls += v[i]; }
  red[tid] = ls; __syncthreads();
  for (int o = 128; o > 0; o >>= 1) {
    if (tid < o) red[tid] += red[tid + o];
    __syncthreads();
  }
  const float inv = 1.f / red[0];
#pragma unroll
  for (int i = 0; i < 4; ++i) dst[tid + 256 * i] = (bf16_t)(v[i] * inv);
}

// ---------- router: logits -> softmax -> top2 -> normalized per-expert weights ----------
__global__ __launch_bounds__(256) void router_kernel(
    const float* __restrict__ h2f, const float* __restrict__ rw,
    float* __restrict__ wgt)
{
  const int s = blockIdx.x, tid = threadIdx.x;
  __shared__ float red[256];
  __shared__ float logits[NEXP];
  for (int e = 0; e < NEXP; ++e) {
    float p = 0.f;
    for (int i = tid; i < Hdim; i += 256)
      p += h2f[(long)s * Hdim + i] * rw[(long)e * Hdim + i];
    red[tid] = p; __syncthreads();
    for (int o = 128; o > 0; o >>= 1) {
      if (tid < o) red[tid] += red[tid + o];
      __syncthreads();
    }
    if (tid == 0) logits[e] = red[0];
    __syncthreads();
  }
  if (tid == 0) {
    int i0 = 0; float l0 = logits[0];
    int i1 = -1; float l1 = -3.4e38f;
    for (int e = 1; e < NEXP; ++e) {
      const float l = logits[e];
      if (l > l0) { l1 = l0; i1 = i0; l0 = l; i0 = e; }
      else if (l > l1) { l1 = l; i1 = e; }
    }
    const float d = expf(l1 - l0);
    const float w0 = 1.f / (1.f + d);
    const float w1 = d / (1.f + d);
    for (int e = 0; e < NEXP; ++e)
      wgt[(long)s * NEXP + e] = (e == i0) ? w0 : ((e == i1) ? w1 : 0.f);
  }
}

// ---------- host-side GEMM dispatch over weight type ----------
template <int EPI>
static inline void launchGemm(dim3 grid, hipStream_t stream,
                              const bf16_t* A, int lda, long aB,
                              const void* W, bool wbf, int ldw, long wB,
                              void* C, int ldc, long cB,
                              int K, float scale,
                              const float* R, int ldr,
                              const float* tw, int tws)
{
  if (wbf)
    gemm_kernel<EPI, bf16_t><<<grid, 256, 0, stream>>>(
        A, lda, aB, (const bf16_t*)W, ldw, wB, C, ldc, cB, K, scale, R, ldr, tw, tws);
  else
    gemm_kernel<EPI, float><<<grid, 256, 0, stream>>>(
        A, lda, aB, (const float*)W, ldw, wB, C, ldc, cB, K, scale, R, ldr, tw, tws);
}

// ---------- host orchestration ----------
extern "C" void kernel_launch(void* const* d_in, const int* in_sizes, int n_in,
                              void* d_out, int out_size, void* d_ws, size_t ws_size,
                              hipStream_t stream)
{
  const int kind[NEXP] = {0, 0, 1, 1, 2, 2, 2, 2};  // 0=quantum 1=creativity 2=general
  const float* x = (const float*)d_in[0];
  const float *ln1w, *ln1b, *ln2w, *ln2b, *qw, *kw, *vw, *ow, *rw;
  const float *egate[NEXP], *eup[NEXP], *edown[NEXP], *ecre[NEXP];
  int einter[NEXP];

  const bool insertionOrder = (n_in > 1 && in_sizes[1] == 1024);
  if (insertionOrder) {
    ln1w = (const float*)d_in[1]; ln1b = (const float*)d_in[2];
    ln2w = (const float*)d_in[3]; ln2b = (const float*)d_in[4];
    qw = (const float*)d_in[5]; kw = (const float*)d_in[6];
    vw = (const float*)d_in[7]; ow = (const float*)d_in[8];
    rw = (const float*)d_in[9];
    int idx = 10;
    for (int e = 0; e < NEXP; ++e) {
      egate[e] = (const float*)d_in[idx++];
      eup[e]   = (const float*)d_in[idx++];
      edown[e] = (const float*)d_in[idx++];
      ecre[e]  = (kind[e] == 1) ? (const float*)d_in[idx++] : nullptr;
      einter[e] = (kind[e] == 0) ? 4096 : 2048;
    }
  } else {
    // pytree-sorted flattening: experts first (each dict key-sorted), then sorted params keys
    int idx = 1;
    for (int e = 0; e < NEXP; ++e) {
      ecre[e]  = (kind[e] == 1) ? (const float*)d_in[idx++] : nullptr;
      edown[e] = (const float*)d_in[idx++];
      egate[e] = (const float*)d_in[idx++];
      eup[e]   = (const float*)d_in[idx++];
      einter[e] = (kind[e] == 0) ? 4096 : 2048;
    }
    kw   = (const float*)d_in[idx++];
    ln1b = (const float*)d_in[idx++]; ln1w = (const float*)d_in[idx++];
    ln2b = (const float*)d_in[idx++]; ln2w = (const float*)d_in[idx++];
    ow   = (const float*)d_in[idx++]; qw   = (const float*)d_in[idx++];
    rw   = (const float*)d_in[idx++]; vw   = (const float*)d_in[idx++];
  }

  // ---- workspace carve: activation buffers (~51 MB) ----
  char* p = (char*)d_ws;
  auto carve = [&](size_t bytes) -> void* {
    void* r = (void*)p;
    p += (bytes + 255) & ~(size_t)255;
    return r;
  };
  bf16_t* h1b = (bf16_t*)carve((size_t)S * Hdim * 2);
  bf16_t* h2b = (bf16_t*)carve((size_t)S * Hdim * 2);
  float*  h2f = (float*) carve((size_t)S * Hdim * 4);
  float*  qf  = (float*) carve((size_t)S * Hdim * 4);
  float*  kf  = (float*) carve((size_t)S * HDh * 4);
  float*  vf  = (float*) carve((size_t)S * HDh * 4);
  bf16_t* qb  = (bf16_t*)carve((size_t)S * Hdim * 2);
  bf16_t* kb  = (bf16_t*)carve((size_t)S * HDh * 2);
  bf16_t* vTb = (bf16_t*)carve((size_t)HDh * S * 2);
  float*  sc  = (float*) carve((size_t)NHEAD * S * S * 4);
  bf16_t* at  = (bf16_t*)carve((size_t)NHEAD * S * S * 2);
  bf16_t* aob = (bf16_t*)carve((size_t)S * Hdim * 2);
  bf16_t* gub = (bf16_t*)carve((size_t)S * 4096 * 2);
  float*  wgt = (float*) carve((size_t)S * NEXP * 4);
  (void)out_size;

  // ---- bf16 weight cache (L2-resident: ~129 MB) if workspace allows ----
  long wElems = 2ll * 1024 * 1024 + 2ll * 256 * 1024 + 1024ll * 1024; // q + k + v + o? (computed below precisely)
  wElems = (long)1048576 + 262144 + 262144 + 1048576;                 // q,k,v,o
  for (int e = 0; e < NEXP; ++e) {
    wElems += 3ll * einter[e] * 1024;
    if (kind[e] == 1) wElems += 1048576;
  }
  const size_t baseBytes = (size_t)(p - (char*)d_ws);
  const bool wbf = (ws_size >= baseBytes + (size_t)wElems * 2 + 4096);

  auto convert = [&](const float* src, long n) -> const void* {
    if (!wbf) return (const void*)src;
    bf16_t* dst = (bf16_t*)carve((size_t)n * 2);
    cvt_bf16_kernel<<<dim3((unsigned)(n / 2048)), 256, 0, stream>>>(src, dst, n);
    return (const void*)dst;
  };

  const void* qwc = convert(qw, 1048576);
  const void* kwc = convert(kw, 262144);
  const void* vwc = convert(vw, 262144);
  const void* owc = convert(ow, 1048576);
  const void* gatec[NEXP]; const void* upc[NEXP];
  const void* downc[NEXP]; const void* crec[NEXP];
  for (int e = 0; e < NEXP; ++e) {
    const long gn = (long)einter[e] * 1024;
    gatec[e] = convert(egate[e], gn);
    upc[e]   = convert(eup[e],   gn);
    downc[e] = convert(edown[e], gn);
    crec[e]  = ecre[e] ? convert(ecre[e], 1048576) : nullptr;
  }

  float* out = (float*)d_out;
  const dim3 blk(256);

  // 1) LN1 -> bf16
  ln_kernel<<<S, blk, 0, stream>>>(x, ln1w, ln1b, h1b, nullptr);

  // 2) q, k, v projections (fp32 out)
  launchGemm<EPI_F32>(dim3(Hdim / 256, S / 64, 1), stream,
      h1b, Hdim, 0, qwc, wbf, Hdim, 0, qf, Hdim, 0, Hdim, 1.f, nullptr, 0, nullptr, 0);
  launchGemm<EPI_F32>(dim3(HDh / 256, S / 64, 1), stream,
      h1b, Hdim, 0, kwc, wbf, Hdim, 0, kf, HDh, 0, Hdim, 1.f, nullptr, 0, nullptr, 0);
  launchGemm<EPI_F32>(dim3(HDh / 256, S / 64, 1), stream,
      h1b, Hdim, 0, vwc, wbf, Hdim, 0, vf, HDh, 0, Hdim, 1.f, nullptr, 0, nullptr, 0);

  // 3) RoPE q/k -> bf16; V -> V^T bf16
  ropev_kernel<<<S, blk, 0, stream>>>(qf, kf, vf, qb, kb, vTb);

  // 4) scores[h] = (q_h @ k^T) / 16   (batched over heads via z; W is bf16 activations)
  gemm_kernel<EPI_F32, bf16_t><<<dim3(S / 256, S / 64, NHEAD), blk, 0, stream>>>(
      qb, Hdim, HDh, kb, HDh, 0, sc, S, (long)S * S, HDh, 0.0625f,
      nullptr, 0, nullptr, 0);

  // 5) softmax rows -> bf16
  softmax_kernel<<<NHEAD * S, blk, 0, stream>>>(sc, at);

  // 6) ao[h] = attn_h @ V  (W = V^T bf16), write bf16 into interleaved [S,H]
  gemm_kernel<EPI_BF16, bf16_t><<<dim3(HDh / 256, S / 64, NHEAD), blk, 0, stream>>>(
      at, S, (long)S * S, vTb, S, 0, aob, Hdim, HDh, S, 1.f,
      nullptr, 0, nullptr, 0);

  // 7) x1 = x + ao @ o_w^T  -> d_out (fp32)
  launchGemm<EPI_RES>(dim3(Hdim / 256, S / 64, 1), stream,
      aob, Hdim, 0, owc, wbf, Hdim, 0, out, Hdim, 0, Hdim, 1.f, x, Hdim, nullptr, 0);

  // 8) LN2 -> bf16 + fp32
  ln_kernel<<<S, blk, 0, stream>>>(out, ln2w, ln2b, h2b, h2f);

  // 9) router weights
  router_kernel<<<S, blk, 0, stream>>>(h2f, rw, wgt);

  // 10) dense MoE, accumulated into d_out (weight==0 for unselected experts)
  for (int e = 0; e < NEXP; ++e) {
    const int inter = einter[e];
    const float qs = (kind[e] == 0) ? 1.1f : 1.f;
    if (wbf)
      gateup_kernel<bf16_t><<<dim3(inter / 256, S / 64, 1), blk, 0, stream>>>(
          h2b, (const bf16_t*)gatec[e], (const bf16_t*)upc[e], gub, inter, Hdim, qs);
    else
      gateup_kernel<float><<<dim3(inter / 256, S / 64, 1), blk, 0, stream>>>(
          h2b, (const float*)gatec[e], (const float*)upc[e], gub, inter, Hdim, qs);
    launchGemm<EPI_DOWN>(dim3(Hdim / 256, S / 64, 1), stream,
        gub, inter, 0, downc[e], wbf, inter, 0, out, Hdim, 0, inter, 1.f,
        nullptr, 0, wgt + e, NEXP);
    if (ecre[e]) {
      launchGemm<EPI_TANH>(dim3(Hdim / 256, S / 64, 1), stream,
          h2b, Hdim, 0, crec[e], wbf, Hdim, 0, out, Hdim, 0, Hdim, 1.f,
          nullptr, 0, wgt + e, NEXP);
    }
  }
}
